// Recurrence_73504070304401
// MI455X (gfx1250) — compile-verified, matrix operand order checked by reading, not run on hardware
//
#include <hip/hip_runtime.h>
#include <hip/hip_bf16.h>

// Shapes from reference
#define T_  32
#define N_  16
#define C_  64
#define L_  64
#define H_  128
#define NL_ 128
#define NA_ 128
#define TOT 323          // 3 + H + NA + L
#define H3  384          // 3*H

typedef __attribute__((ext_vector_type(16))) _Float16 v16h;
typedef __attribute__((ext_vector_type(8)))  float    v8f;

union V16H { v16h v; _Float16 h[16]; };
union V8F  { v8f  v; float    f[8];  };

// Fast transcendental helpers: HW rcp/exp pipes instead of IEEE div / libm.
__device__ __forceinline__ float fast_rcp(float x) { return __builtin_amdgcn_rcpf(x); }
__device__ __forceinline__ float sigf(float x) { return fast_rcp(1.0f + __expf(-x)); }
__device__ __forceinline__ float tanh_fast(float x) {
    float cx = fminf(fmaxf(x, -15.f), 15.f);   // saturate; avoids inf*0
    float e  = __expf(2.f * cx);
    return (e - 1.f) * fast_rcp(e + 1.f);
}

// ---------------------------------------------------------------------------
// Kernel A: GI4[dir][l][n][hc] = float4(pre_r, pre_z, pre_n, -)
//   pre_g = bih[g*H+hc] + (g<2 ? bhh[g*H+hc] : 0)
//           + dot(emb[lines_obs[0,n,l]], Wih[g*H+hc,:])
// Gates interleaved into a padded float4 so the recurrence reads ONE b128
// per element instead of 3 scattered b32. r/z hidden biases folded here.
// ---------------------------------------------------------------------------
__global__ __launch_bounds__(256) void gi_kernel(
    const int* __restrict__ lines_obs, const float* __restrict__ emb,
    const float* __restrict__ Wih_f, const float* __restrict__ bih_f,
    const float* __restrict__ Wih_b, const float* __restrict__ bih_b,
    const float* __restrict__ bhh_f, const float* __restrict__ bhh_b,
    float* __restrict__ GI4)           // [2][64][16][128][4]
{
    int idx = blockIdx.x * blockDim.x + threadIdx.x;
    const int total = 2 * L_ * N_ * H3;
    if (idx >= total) return;
    int j   = idx % H3;
    int n   = (idx / H3) % N_;
    int l   = (idx / (H3 * N_)) % L_;
    int dir = idx / (H3 * N_ * L_);
    const float* Wih = dir ? Wih_b : Wih_f;
    const float* bih = dir ? bih_b : bih_f;
    const float* bhh = dir ? bhh_b : bhh_f;
    int tok = lines_obs[n * L_ + l];        // lines_obs[0][n][l]
    const float* x = emb + tok * H_;
    const float* w = Wih + j * H_;
    float acc = 0.f;
#pragma unroll 4
    for (int k = 0; k < H_; ++k) acc += x[k] * w[k];
    acc += bih[j];
    if (j < 2 * H_) acc += bhh[j];          // fold r/z hidden bias
    int g  = j / H_;                        // gate: 0=r,1=z,2=n
    int hc = j % H_;
    size_t base = ((((size_t)dir * L_ + l) * N_ + n) * H_ + hc) * 4;
    GI4[base + g] = acc;
    if (g == 2) GI4[base + 3] = 0.f;        // init pad lane
}

// ---------------------------------------------------------------------------
// Kernel B: bidirectional rotated-GRU recurrence.
//   128 blocks = (dir, r). Block owns 16 batch rows (n) of hidden state.
//   Per step: GI prefetch (b128, hidden behind GEMM) -> gh = h @ Whh^T via
//   v_wmma_f32_16x16x32_f16 -> gates -> h update -> emit S[r][s][n][dir][:].
// Whh B-fragments and the n-gate bias are loop-invariant -> kept in VGPRs.
// ---------------------------------------------------------------------------
__global__ __launch_bounds__(256) void gru_bidir_kernel(
    const float* __restrict__ GI4,    // [2][L][N][H][4]
    const float* __restrict__ Whh_f, const float* __restrict__ Whh_b, // [3H][H]
    const float* __restrict__ bhh_f, const float* __restrict__ bhh_b, // [3H]
    float* __restrict__ S)            // [L][L][N][2][H]
{
    const int blk = blockIdx.x;       // 0..127
    const int dir = blk >> 6;
    const int r   = blk & 63;
    const float* Whh  = dir ? Whh_b : Whh_f;
    const float* bhhn = (dir ? bhh_b : bhh_f) + 2 * H_;   // n-gate bias only
    const float4* GId = (const float4*)GI4 + (size_t)dir * (L_ * N_ * H_);

    __shared__ _Float16 h16[N_ * H_];   // f16 hidden (WMMA A staging)
    __shared__ float    hs [N_ * H_];   // f32 hidden
    __shared__ float    ghs[N_ * H3];   // gate pre-activations

    const int tid  = threadIdx.x;
    const int wave = tid >> 5;          // 8 waves (wave32)
    const int lane = tid & 31;
    const int m    = lane & 15;         // row within 16x16 tile
    const int hi   = (lane >= 16) ? 8 : 0;

    // update-phase ownership: 8 CONTIGUOUS elements -> b128 LDS/global ops
    const int ebase = tid * 8;          // 0..2047
    const int urow  = ebase >> 7;       // n
    const int uhc   = ebase & 127;      // hidden col base (multiple of 8)
    float bn[8];
#pragma unroll
    for (int i = 0; i < 8; ++i) bn[i] = bhhn[uhc + i];   // loop-invariant

    for (int i = tid; i < N_ * H_; i += 256) { hs[i] = 0.f; h16[i] = (_Float16)0.f; }

    // Preload invariant B fragments: 3 column tiles x 4 k-chunks per wave.
    V16H Bfrag[3][4];
#pragma unroll
    for (int j = 0; j < 3; ++j) {
        int col = (wave * 3 + j) * 16 + m;          // global output column
#pragma unroll
        for (int kc = 0; kc < 4; ++kc) {
            int kb = kc * 32 + hi;
#pragma unroll
            for (int i = 0; i < 16; ++i) {
                int k = kb + (i < 8 ? i : i + 8);
                Bfrag[j][kc].h[i] = (_Float16)Whh[col * H_ + k];
            }
        }
    }
    __syncthreads();

    for (int step = 0; step < L_; ++step) {
        const int s = dir ? (63 - step) : step;     // output position
        const int l = (s - r) & 63;                 // rotated input index

        // --- GI prefetch: independent of GEMM, latency hides behind WMMA ---
        const float4* gil = GId + (size_t)l * (N_ * H_);
        float4 gv[8];
#pragma unroll
        for (int i = 0; i < 8; ++i) gv[i] = gil[ebase + i];

        // --- GEMM: gh = h @ Whh^T  (16 x 384, K=128) ---
        V8F acc[3];
#pragma unroll
        for (int j = 0; j < 3; ++j)
#pragma unroll
            for (int v = 0; v < 8; ++v) acc[j].f[v] = 0.f;

#pragma unroll
        for (int kc = 0; kc < 4; ++kc) {
            const int kb = kc * 32 + hi;
            V16H a;
            const _Float16* p0 = &h16[m * H_ + kb];
            const _Float16* p1 = &h16[m * H_ + kb + 16];
#pragma unroll
            for (int i = 0; i < 8; ++i) { a.h[i] = p0[i]; a.h[8 + i] = p1[i]; }
#pragma unroll
            for (int j = 0; j < 3; ++j)
                acc[j].v = __builtin_amdgcn_wmma_f32_16x16x32_f16(
                    false, a.v, false, Bfrag[j][kc].v, (short)0, acc[j].v, false, false);
        }
        // store C tiles: VGPR v -> row (hi? v+8 : v), col = tile_base + (lane&15)
#pragma unroll
        for (int j = 0; j < 3; ++j) {
            int colbase = (wave * 3 + j) * 16;
#pragma unroll
            for (int v = 0; v < 8; ++v)
                ghs[(hi + v) * H3 + colbase + m] = acc[j].f[v];
        }
        __syncthreads();

        // --- GRU gates + hidden update + emit S (contiguous b128 traffic) ---
        const float* ghr_p = &ghs[urow * H3 + uhc];
        const float* ghz_p = ghr_p + H_;
        const float* ghn_p = ghr_p + 2 * H_;
        float* Sp = S + ((((size_t)r * L_ + s) * N_ + urow) * 2 + dir) * H_ + uhc;
#pragma unroll
        for (int i = 0; i < 8; ++i) {
            float4 g = gv[i];
            float rg = sigf(g.x + ghr_p[i]);
            float zg = sigf(g.y + ghz_p[i]);
            float ng = tanh_fast(g.z + rg * (ghn_p[i] + bn[i]));
            float hv = (1.f - zg) * ng + zg * hs[ebase + i];
            hs[ebase + i]  = hv;
            h16[ebase + i] = (_Float16)hv;
            Sp[i] = hv;
        }
        __syncthreads();
    }
}

// ---------------------------------------------------------------------------
// Generic dense layer: Y[row,j] = act(b[j] + sum_k X[row,k] * W[k,j])
// ---------------------------------------------------------------------------
__global__ __launch_bounds__(256) void mlp_kernel(
    const float* __restrict__ X, const float* __restrict__ W,
    const float* __restrict__ b, float* __restrict__ Y,
    int rows, int K, int N, int do_relu)
{
    int idx = blockIdx.x * blockDim.x + threadIdx.x;
    if (idx >= rows * N) return;
    int row = idx / N, j = idx % N;
    const float* x = X + row * K;
    float acc = b[j];
    for (int k = 0; k < K; ++k) acc += x[k] * W[k * N + j];
    Y[idx] = (do_relu && acc < 0.f) ? 0.f : acc;
}

// all_inputs = concat(cond, float(lines_obs))  -> [512][128]
__global__ __launch_bounds__(256) void cat_kernel(
    const float* __restrict__ cond, const int* __restrict__ lines_obs,
    float* __restrict__ Y)
{
    int idx = blockIdx.x * blockDim.x + threadIdx.x;
    if (idx >= T_ * N_ * (C_ + L_)) return;
    int c  = idx % (C_ + L_);
    int rn = idx / (C_ + L_);
    Y[idx] = (c < C_) ? cond[rn * C_ + c] : (float)lines_obs[rn * L_ + (c - C_)];
}

// ---------------------------------------------------------------------------
// Attention: a_probs[t,n,:] = softmax_k( dot(S[pt, s, n, d, :], q[t,n,:]) ),
// k = 2*s + d.  One block per (t,n), 128 threads, LDS softmax reduction.
// ---------------------------------------------------------------------------
__global__ __launch_bounds__(128) void attn_kernel(
    const float* __restrict__ S, const float* __restrict__ q_all,
    const int* __restrict__ given_p, float* __restrict__ out)
{
    int rn = blockIdx.x;                 // t*16 + n
    int n  = rn & 15;
    int pt = given_p[rn];
    int k  = threadIdx.x;                // 0..127
    int s  = k >> 1, d = k & 1;

    __shared__ float qs[H_];
    __shared__ float buf[NA_];
    qs[k] = q_all[rn * H_ + k];
    __syncthreads();

    const float* key = S + ((((size_t)pt * L_ + s) * N_ + n) * 2 + d) * H_;
    float acc = 0.f;
#pragma unroll 4
    for (int h = 0; h < H_; ++h) acc += key[h] * qs[h];

    buf[k] = acc; __syncthreads();
    for (int off = 64; off > 0; off >>= 1) {
        if (k < off) buf[k] = fmaxf(buf[k], buf[k + off]);
        __syncthreads();
    }
    float mx = buf[0]; __syncthreads();
    float ex = __expf(acc - mx);
    buf[k] = ex; __syncthreads();
    for (int off = 64; off > 0; off >>= 1) {
        if (k < off) buf[k] += buf[k + off];
        __syncthreads();
    }
    out[(size_t)rn * TOT + 3 + H_ + k] = ex * fast_rcp(buf[0]);
}

// ---------------------------------------------------------------------------
// gc GRU scan over T (single persistent block) + action MLP + p_probs softmax
// ---------------------------------------------------------------------------
__global__ __launch_bounds__(256) void gc_kernel(
    const float* __restrict__ hxs, const int* __restrict__ given_a,
    const float* __restrict__ aemb,
    const float* __restrict__ Wih, const float* __restrict__ Whh,
    const float* __restrict__ bih, const float* __restrict__ bhh,
    const float* __restrict__ aW0, const float* __restrict__ ab0,
    const float* __restrict__ aW1, const float* __restrict__ ab1,
    const float* __restrict__ aWc, const float* __restrict__ abc,
    float* __restrict__ out)
{
    __shared__ float h [N_ * H_];   // 8 KB
    __shared__ float gi[N_ * H3];   // 24 KB (reused for x2 / logits)
    __shared__ float gh[N_ * H3];   // 24 KB (reused for x1)
    const int tid = threadIdx.x;

    for (int i = tid; i < N_ * H_; i += 256)
        h[i] = hxs[(i >> 7) * TOT + 3 + (i & 127)];   // h0 from rnn_hxs
    __syncthreads();

    for (int t = 0; t < T_; ++t) {
        // gate pre-activations
        for (int e = tid; e < N_ * H3; e += 256) {
            int n = e / H3, j = e % H3;
            int ap = (t == 0) ? (int)hxs[n * TOT] : given_a[(t - 1) * N_ + n];
            const float* x  = aemb + ap * H_;
            const float* wi = Wih + j * H_;
            const float* wh = Whh + j * H_;
            float ai = bih[j], ah = bhh[j];
            for (int k = 0; k < H_; ++k) { ai += x[k] * wi[k]; ah += h[n * H_ + k] * wh[k]; }
            gi[e] = ai; gh[e] = ah;
        }
        __syncthreads();
        // hidden update
        for (int e = tid; e < N_ * H_; e += 256) {
            int n = e >> 7, hc = e & 127;
            float rg = sigf(gi[n * H3 + hc]       + gh[n * H3 + hc]);
            float zg = sigf(gi[n * H3 + H_ + hc]  + gh[n * H3 + H_ + hc]);
            float ng = tanh_fast(gi[n * H3 + 2 * H_ + hc] + rg * gh[n * H3 + 2 * H_ + hc]);
            h[e] = (1.f - zg) * ng + zg * h[e];
        }
        __syncthreads();
        // x1 = relu(h @ aW0 + ab0) -> gh
        for (int e = tid; e < N_ * H_; e += 256) {
            int n = e >> 7, j = e & 127;
            float acc = ab0[j];
            for (int k = 0; k < H_; ++k) acc += h[n * H_ + k] * aW0[k * H_ + j];
            gh[e] = acc > 0.f ? acc : 0.f;
        }
        __syncthreads();
        // x2 = relu(x1 @ aW1 + ab1) -> gi
        for (int e = tid; e < N_ * H_; e += 256) {
            int n = e >> 7, j = e & 127;
            float acc = ab1[j];
            for (int k = 0; k < H_; ++k) acc += gh[n * H_ + k] * aW1[k * H_ + j];
            gi[e] = acc > 0.f ? acc : 0.f;
        }
        __syncthreads();
        // logits = x2 @ aWc + abc -> gi[N_*H_ ...] region (separate from x2)
        float* logits = gi + N_ * H_;      // 16*64 floats fits in remaining gi
        for (int e = tid; e < N_ * L_; e += 256) {
            int n = e / L_, c = e % L_;
            float acc = abc[c];
            for (int k = 0; k < H_; ++k) acc += gi[n * H_ + k] * aWc[k * L_ + c];
            logits[e] = acc;
        }
        __syncthreads();
        // softmax per row, write p_probs
        if (tid < N_) {
            int n = tid;
            float mx = -1e30f;
            for (int c = 0; c < L_; ++c) mx = fmaxf(mx, logits[n * L_ + c]);
            float sum = 0.f;
            for (int c = 0; c < L_; ++c) sum += __expf(logits[n * L_ + c] - mx);
            float inv = fast_rcp(sum);
            size_t base = ((size_t)t * N_ + n) * TOT + 3 + H_ + NA_;
            for (int c = 0; c < L_; ++c)
                out[base + c] = __expf(logits[n * L_ + c] - mx) * inv;
        }
        __syncthreads();
    }
}

// ---------------------------------------------------------------------------
// fill cols 0..130: given_a, given_p, v_all, h_rep (= h0 broadcast)
// ---------------------------------------------------------------------------
__global__ __launch_bounds__(256) void fill_kernel(
    const int* __restrict__ ga, const int* __restrict__ gp,
    const float* __restrict__ v2, const float* __restrict__ hxs,
    float* __restrict__ out)
{
    int idx = blockIdx.x * blockDim.x + threadIdx.x;
    if (idx >= T_ * N_ * 131) return;
    int rn = idx / 131, c = idx % 131;
    float val;
    if      (c == 0) val = (float)ga[rn];
    else if (c == 1) val = (float)gp[rn];
    else if (c == 2) val = v2[rn];
    else             val = hxs[(rn & 15) * TOT + c];   // h0[n][c-3] at hxs offset 3+(c-3)=c
    out[(size_t)rn * TOT + c] = val;
}

// second output = hx_out[-1:]
__global__ __launch_bounds__(256) void tail_kernel(float* __restrict__ out)
{
    int idx = blockIdx.x * blockDim.x + threadIdx.x;
    if (idx >= N_ * TOT) return;
    out[(size_t)T_ * N_ * TOT + idx] = out[(size_t)(T_ - 1) * N_ * TOT + idx];
}

// ---------------------------------------------------------------------------
extern "C" void kernel_launch(void* const* d_in, const int* in_sizes, int n_in,
                              void* d_out, int out_size, void* d_ws, size_t ws_size,
                              hipStream_t stream) {
    const float* cond      = (const float*)d_in[0];
    const int*   lines_obs = (const int*)  d_in[1];
    const int*   given_a   = (const int*)  d_in[2];
    const int*   given_p   = (const int*)  d_in[3];
    const float* rnn_hxs   = (const float*)d_in[4];
    const float* emb       = (const float*)d_in[5];
    const float* aemb      = (const float*)d_in[6];
    const float* tg_Wih_f  = (const float*)d_in[7];
    const float* tg_Whh_f  = (const float*)d_in[8];
    const float* tg_Wih_b  = (const float*)d_in[9];
    const float* tg_Whh_b  = (const float*)d_in[10];
    const float* gc_Wih    = (const float*)d_in[11];
    const float* gc_Whh    = (const float*)d_in[12];
    const float* tg_bih_f  = (const float*)d_in[13];
    const float* tg_bhh_f  = (const float*)d_in[14];
    const float* tg_bih_b  = (const float*)d_in[15];
    const float* tg_bhh_b  = (const float*)d_in[16];
    const float* gc_bih    = (const float*)d_in[17];
    const float* gc_bhh    = (const float*)d_in[18];
    const float* f_W0 = (const float*)d_in[19]; const float* f_b0 = (const float*)d_in[20];
    const float* f_W1 = (const float*)d_in[21]; const float* f_b1 = (const float*)d_in[22];
    const float* f_W2 = (const float*)d_in[23]; const float* f_b2 = (const float*)d_in[24];
    const float* c_W0 = (const float*)d_in[25]; const float* c_b0 = (const float*)d_in[26];
    const float* c_W1 = (const float*)d_in[27]; const float* c_b1 = (const float*)d_in[28];
    const float* c_W2 = (const float*)d_in[29]; const float* c_b2 = (const float*)d_in[30];
    const float* a_W0 = (const float*)d_in[31]; const float* a_b0 = (const float*)d_in[32];
    const float* a_W1 = (const float*)d_in[33]; const float* a_b1 = (const float*)d_in[34];
    const float* a_Wc = (const float*)d_in[35]; const float* a_bc = (const float*)d_in[36];

    float* out = (float*)d_out;
    float* ws  = (float*)d_ws;

    // workspace layout (floats)
    float* GI4 = ws;                          // [2][64][16][128][4] = 1048576
    float* S   = GI4 + 2 * L_ * N_ * H_ * 4;  // [64][64][16][2][128] = 16777216
    float* q0  = S + (size_t)L_ * L_ * N_ * 2 * H_;
    float* q1  = q0 + T_ * N_ * H_;
    float* q2  = q1 + T_ * N_ * H_;           // q_all
    float* cat = q2 + T_ * N_ * H_;           // [512][128]
    float* v0  = cat + T_ * N_ * (C_ + L_);
    float* v1  = v0 + T_ * N_ * H_;
    float* v2  = v1 + T_ * N_ * H_;           // [512]

    const int RN = T_ * N_;                   // 512

    // 1) input-side GRU pre-activations (hoisted, gate-interleaved, biases folded)
    gi_kernel<<<(2 * L_ * N_ * H3 + 255) / 256, 256, 0, stream>>>(
        lines_obs, emb, tg_Wih_f, tg_bih_f, tg_Wih_b, tg_bih_b,
        tg_bhh_f, tg_bhh_b, GI4);

    // 2) bidirectional rotated-GRU recurrence (WMMA) -> S
    gru_bidir_kernel<<<128, 256, 0, stream>>>(
        GI4, tg_Whh_f, tg_Whh_b, tg_bhh_f, tg_bhh_b, S);

    // 3) q_all MLP chain
    mlp_kernel<<<(RN * H_ + 255) / 256, 256, 0, stream>>>(cond, f_W0, f_b0, q0, RN, C_, H_, 1);
    mlp_kernel<<<(RN * H_ + 255) / 256, 256, 0, stream>>>(q0,   f_W1, f_b1, q1, RN, H_, H_, 1);
    mlp_kernel<<<(RN * H_ + 255) / 256, 256, 0, stream>>>(q1,   f_W2, f_b2, q2, RN, H_, H_, 1);

    // 4) v_all MLP chain
    cat_kernel<<<(RN * (C_ + L_) + 255) / 256, 256, 0, stream>>>(cond, lines_obs, cat);
    mlp_kernel<<<(RN * H_ + 255) / 256, 256, 0, stream>>>(cat, c_W0, c_b0, v0, RN, C_ + L_, H_, 1);
    mlp_kernel<<<(RN * H_ + 255) / 256, 256, 0, stream>>>(v0,  c_W1, c_b1, v1, RN, H_, H_, 1);
    mlp_kernel<<<(RN + 255) / 256, 256, 0, stream>>>(v1,  c_W2, c_b2, v2, RN, H_, 1, 0);

    // 5) misc output columns 0..130
    fill_kernel<<<(RN * 131 + 255) / 256, 256, 0, stream>>>(given_a, given_p, v2, rnn_hxs, out);

    // 6) attention -> a_probs (cols 131..258)
    attn_kernel<<<RN, 128, 0, stream>>>(S, q2, given_p, out);

    // 7) gc GRU scan + action MLP -> p_probs (cols 259..322)
    gc_kernel<<<1, 256, 0, stream>>>(rnn_hxs, given_a, aemb,
                                     gc_Wih, gc_Whh, gc_bih, gc_bhh,
                                     a_W0, a_b0, a_W1, a_b1, a_Wc, a_bc, out);

    // 8) hx_out[-1:] tail
    tail_kernel<<<(N_ * TOT + 255) / 256, 256, 0, stream>>>(out);
}
